// SigLipLoss_38895223833296
// MI455X (gfx1250) — matrix operand order repaired and checked by reading, not compiled, not run on hardware
//
#include <hip/hip_runtime.h>

// ---------------------------------------------------------------------------
// SigLIP loss on MI455X (gfx1250):
//   k1: L2-normalize rows + convert f32 -> bf16 (both matrices -> workspace)
//   k2: 128x128-tiled bf16 WMMA GEMM (16x16x32), per-wave 32x64 register tile,
//       double-buffered LDS staged with GLOBAL_LOAD_ASYNC_TO_LDS_B128,
//       fused stable log-sigmoid loss + f64 atomic accumulate
//   k3: finalize: out = -(sum / N^2)
// ---------------------------------------------------------------------------

#define N_ROWS 16384
#define DIM    1024
#define BLK    128    // workgroup output tile edge
#define KSTEP  32     // WMMA bf16 K per step
#define LDS_STRIDE 40 // bf16 elems per LDS tile row (32 data + 8 pad = 80B)

typedef __bf16 bf16;
typedef __bf16 v4bf  __attribute__((ext_vector_type(4)));
typedef __bf16 v8bf  __attribute__((ext_vector_type(8)));
typedef __bf16 v16bf __attribute__((ext_vector_type(16)));
typedef float  v8f   __attribute__((ext_vector_type(8)));

// -------------------------- kernel 1: normalize ----------------------------
__global__ __launch_bounds__(256) void siglip_normalize_bf16(
    const float* __restrict__ img, const float* __restrict__ txt,
    bf16* __restrict__ imgB, bf16* __restrict__ txtB)
{
    __shared__ float red[256];
    int row = blockIdx.x;
    const float* src;
    bf16* dst;
    if (row < N_ROWS) {
        src = img + (size_t)row * DIM;
        dst = imgB + (size_t)row * DIM;
    } else {
        row -= N_ROWS;
        src = txt + (size_t)row * DIM;
        dst = txtB + (size_t)row * DIM;
    }
    const int t = threadIdx.x;
    float4 v = reinterpret_cast<const float4*>(src)[t]; // 256 thr * 4 = 1024
    float ss = v.x * v.x + v.y * v.y + v.z * v.z + v.w * v.w;
    red[t] = ss;
    __syncthreads();
    #pragma unroll
    for (int s = 128; s > 0; s >>= 1) {
        if (t < s) red[t] += red[t + s];
        __syncthreads();
    }
    float inv = 1.0f / fmaxf(sqrtf(red[0]), 1e-12f);
    v4bf o;
    o[0] = (bf16)(v.x * inv);
    o[1] = (bf16)(v.y * inv);
    o[2] = (bf16)(v.z * inv);
    o[3] = (bf16)(v.w * inv);
    *reinterpret_cast<v4bf*>(dst + 4 * t) = o;
}

// ------------------- kernel 2: WMMA GEMM + fused loss ----------------------
// Async DMA of one 16B chunk: LDS[ldsOff] <- *gptr (tracked by ASYNCcnt)
#define ASYNC_CP16(ldsOff, gptr)                                              \
    asm volatile("global_load_async_to_lds_b128 %0, %1, off"                  \
                 :: "v"(ldsOff), "v"(gptr) : "memory")
#define WAIT_ASYNC0() asm volatile("s_wait_asynccnt 0" ::: "memory")

__global__ __launch_bounds__(256) void siglip_gemm_loss(
    const bf16* __restrict__ A,          // normalized img [N, D] bf16
    const bf16* __restrict__ B,          // normalized txt [N, D] bf16
    const float* __restrict__ logit_scale,
    const float* __restrict__ logit_bias,
    double* __restrict__ acc)
{
    __shared__ __align__(16) bf16 ldsA[2][BLK * LDS_STRIDE]; // double buffered
    __shared__ __align__(16) bf16 ldsB[2][BLK * LDS_STRIDE];
    __shared__ float red[256];

    const int t       = threadIdx.x;
    const int wave    = t >> 5;          // 8 waves of 32
    const int lane    = t & 31;
    const int laneLow = lane & 15;
    const int g       = lane >> 4;       // half-group (ISA bf16 frag layout)
    const int waveRow = wave >> 1;       // 4 row groups  (32 rows each)
    const int waveCol = wave & 1;        // 2 col groups  (64 cols each)

    const int rowBase = blockIdx.x * BLK;
    const int colBase = blockIdx.y * BLK;

    const float scale = __expf(logit_scale[0]);
    const float bias  = logit_bias[0];

    // staging: 256 threads x 2 chunks of 16B per matrix per K-step
    const int r0 = t >> 2,         c0 = t & 3;          // chunk 0..255
    const int r1 = (t + 256) >> 2, c1 = (t + 256) & 3;  // chunk 256..511

    const bf16* gA0 = A + (size_t)(rowBase + r0) * DIM + c0 * 8;
    const bf16* gA1 = A + (size_t)(rowBase + r1) * DIM + c1 * 8;
    const bf16* gB0 = B + (size_t)(colBase + r0) * DIM + c0 * 8;
    const bf16* gB1 = B + (size_t)(colBase + r1) * DIM + c1 * 8;

    unsigned aOff0[2], aOff1[2], bOff0[2], bOff1[2];
    #pragma unroll
    for (int b = 0; b < 2; ++b) {
        aOff0[b] = (unsigned)(uintptr_t)&ldsA[b][r0 * LDS_STRIDE + c0 * 8];
        aOff1[b] = (unsigned)(uintptr_t)&ldsA[b][r1 * LDS_STRIDE + c1 * 8];
        bOff0[b] = (unsigned)(uintptr_t)&ldsB[b][r0 * LDS_STRIDE + c0 * 8];
        bOff1[b] = (unsigned)(uintptr_t)&ldsB[b][r1 * LDS_STRIDE + c1 * 8];
    }

    v8f accv[2][4] = {};  // per-wave 32x64 tile: 2 A-blocks x 4 B-blocks

    // prologue: DMA first K-slab into buffer 0
    ASYNC_CP16(aOff0[0], gA0);
    ASYNC_CP16(aOff1[0], gA1);
    ASYNC_CP16(bOff0[0], gB0);
    ASYNC_CP16(bOff1[0], gB1);
    WAIT_ASYNC0();
    __syncthreads();

    int p = 0;
    #pragma unroll 2
    for (int k0 = 0; k0 < DIM; k0 += KSTEP) {
        // DMA next K-slab into the other buffer while we compute
        if (k0 + KSTEP < DIM) {
            const int kn = k0 + KSTEP;
            ASYNC_CP16(aOff0[p ^ 1], gA0 + kn);
            ASYNC_CP16(aOff1[p ^ 1], gA1 + kn);
            ASYNC_CP16(bOff0[p ^ 1], gB0 + kn);
            ASYNC_CP16(bOff1[p ^ 1], gB1 + kn);
        }

        const bf16* lA = ldsA[p];
        const bf16* lB = ldsB[p];

        // A fragments (16x32 each): lanes 0-15 -> K 0..7 & 16..23; +8 for hi
        v16bf afrag[2];
        #pragma unroll
        for (int ar = 0; ar < 2; ++ar) {
            const bf16* aRow = lA + (waveRow * 32 + ar * 16 + laneLow) * LDS_STRIDE;
            v8bf alo = *reinterpret_cast<const v8bf*>(aRow + g * 8);
            v8bf ahi = *reinterpret_cast<const v8bf*>(aRow + 16 + g * 8);
            afrag[ar] = __builtin_shufflevector(
                alo, ahi, 0, 1, 2, 3, 4, 5, 6, 7, 8, 9, 10, 11, 12, 13, 14, 15);
        }

        #pragma unroll
        for (int cb = 0; cb < 4; ++cb) {
            // B fragment (32x16): lane = column, K 0..7 & 16..23; +8 for hi
            const bf16* bRow =
                lB + (waveCol * 64 + cb * 16 + laneLow) * LDS_STRIDE + g * 8;
            v8bf blo = *reinterpret_cast<const v8bf*>(bRow);
            v8bf bhi = *reinterpret_cast<const v8bf*>(bRow + 16);
            v16bf bfrag = __builtin_shufflevector(
                blo, bhi, 0, 1, 2, 3, 4, 5, 6, 7, 8, 9, 10, 11, 12, 13, 14, 15);
            #pragma unroll
            for (int ar = 0; ar < 2; ++ar) {
                accv[ar][cb] = __builtin_amdgcn_wmma_f32_16x16x32_bf16(
                    false, afrag[ar], false, bfrag, (short)0, accv[ar][cb],
                    false, false);
            }
        }

        WAIT_ASYNC0();      // my DMA for next slab done
        __syncthreads();    // everyone done reading current slab + DMAs visible
        p ^= 1;
    }

    // fused epilogue: logit = scale*dot + bias; label = +1 diag / -1 off;
    // log_sigmoid(z) = min(z,0) - log1p(exp(-|z|))   (numerically stable)
    float lsum = 0.0f;
    #pragma unroll
    for (int ar = 0; ar < 2; ++ar) {
        const int rBase = rowBase + waveRow * 32 + ar * 16 + g * 8;
        #pragma unroll
        for (int cb = 0; cb < 4; ++cb) {
            const int col = colBase + waveCol * 64 + cb * 16 + laneLow;
            #pragma unroll
            for (int v = 0; v < 8; ++v) {
                const int row = rBase + v;
                float x = fmaf(scale, accv[ar][cb][v], bias);
                float z = (row == col) ? x : -x;
                lsum += fminf(z, 0.0f) - log1pf(__expf(-fabsf(z)));
            }
        }
    }

    red[t] = lsum;
    __syncthreads();
    #pragma unroll
    for (int s = 128; s > 0; s >>= 1) {
        if (t < s) red[t] += red[t + s];
        __syncthreads();
    }
    if (t == 0) atomicAdd(acc, (double)red[0]);
}

// -------------------------- kernel 3: finalize -----------------------------
__global__ void siglip_finalize(const double* __restrict__ acc,
                                float* __restrict__ out)
{
    out[0] = (float)(-acc[0] / ((double)N_ROWS * (double)N_ROWS));
}

// ---------------------------------------------------------------------------
extern "C" void kernel_launch(void* const* d_in, const int* in_sizes, int n_in,
                              void* d_out, int out_size, void* d_ws, size_t ws_size,
                              hipStream_t stream) {
    (void)in_sizes; (void)n_in; (void)out_size; (void)ws_size;
    const float* img = (const float*)d_in[0];
    const float* txt = (const float*)d_in[1];
    const float* lsc = (const float*)d_in[2];
    const float* lbi = (const float*)d_in[3];
    float* out = (float*)d_out;

    char* ws = (char*)d_ws;
    double* acc = (double*)ws;                                   // 8 B
    bf16* imgB = (bf16*)(ws + 256);                              // 32 MB
    bf16* txtB = (bf16*)(ws + 256 + (size_t)N_ROWS * DIM * 2);   // 32 MB

    hipMemsetAsync(acc, 0, sizeof(double), stream);

    siglip_normalize_bf16<<<2 * N_ROWS, 256, 0, stream>>>(img, txt, imgB, txtB);

    dim3 grid(N_ROWS / BLK, N_ROWS / BLK); // 128 x 128 workgroups
    siglip_gemm_loss<<<grid, 256, 0, stream>>>(imgB, txtB, lsc, lbi, acc);

    siglip_finalize<<<1, 1, 0, stream>>>(acc, out);
}
